// STNBLOCK_3556232921203
// MI455X (gfx1250) — matrix operand order repaired and checked by reading, not compiled, not run on hardware
//
#include <hip/hip_runtime.h>
#include <hip/hip_bf16.h>

typedef __attribute__((ext_vector_type(16))) _Float16 v16h;
typedef __attribute__((ext_vector_type(8)))  float    v8f;

#define NB   8
#define C    256
#define Himg 64
#define Wimg 64
#define HW   4096            // 64*64
#define KTAP 9
#define KTOT (C*KTAP)        // 2304
#define KC   32              // K per WMMA
#define NKC  (KTOT/KC)       // 72 chunks; each chunk = one tap, 32 channels
#define OC   256
#define NT   (OC/16)         // 16 n-tiles
#define NTW  16              // n-tiles per wave (full A-fragment reuse)
#define MTOT (NB*HW)         // 32768 pixels
#define MT   (MTOT/16)       // 2048 m-tiles
#define CHUNKH 8192          // halves per K-chunk of Bpack (16 KB)
#define EPSV 1e-5f

// ---------------- Kernel 1: offset conv + bilinear metadata ----------------
__global__ void k_offsets(const float* __restrict__ x,
                          const float* __restrict__ w_tm,
                          const float* __restrict__ b_tm,
                          int4*  __restrict__ midx,   // [MTOT*9]
                          float4* __restrict__ mwgt)  // [MTOT*9]
{
    __shared__ float sw[4*C*9];
    int tid = threadIdx.x;
    #pragma unroll
    for (int i = 0; i < 36; ++i) sw[tid*36 + i] = w_tm[tid*36 + i];
    __syncthreads();

    int p  = blockIdx.x * 256 + tid;       // pixel id
    int n  = p >> 12;
    int hw = p & 4095;
    int h  = hw >> 6;
    int w  = hw & 63;

    float t0 = b_tm[0], t1 = b_tm[1], t2 = b_tm[2], t3 = b_tm[3];
    const float* xn = x + (size_t)n * C * HW;
    for (int c = 0; c < C; ++c) {
        const float* xc = xn + (size_t)c * HW;
        #pragma unroll
        for (int ky = 0; ky < 3; ++ky) {
            int yy = h + ky - 1;
            #pragma unroll
            for (int kx = 0; kx < 3; ++kx) {
                int xx = w + kx - 1;
                float v = (yy >= 0 && yy < Himg && xx >= 0 && xx < Wimg)
                            ? xc[yy*Wimg + xx] : 0.0f;
                int wi = c*9 + ky*3 + kx;
                t0 = fmaf(v, sw[0*C*9 + wi], t0);
                t1 = fmaf(v, sw[1*C*9 + wi], t1);
                t2 = fmaf(v, sw[2*C*9 + wi], t2);
                t3 = fmaf(v, sw[3*C*9 + wi], t3);
            }
        }
    }
    #pragma unroll
    for (int k = 0; k < KTAP; ++k) {
        float dy = (float)(k/3 - 1);
        float dx = (float)(k%3 - 1);
        float offy = t0*dy + t1*dx - dy;
        float offx = t2*dy + t3*dx - dx;
        float py = (float)h + dy + offy;
        float px = (float)w + dx + offx;
        float y0 = floorf(py), x0 = floorf(px);
        float wy1 = py - y0,   wx1 = px - x0;
        float cw[4] = { (1.f-wy1)*(1.f-wx1), (1.f-wy1)*wx1, wy1*(1.f-wx1), wy1*wx1 };
        float cy[4] = { y0, y0, y0+1.f, y0+1.f };
        float cx[4] = { x0, x0+1.f, x0, x0+1.f };
        int   ci[4];
        #pragma unroll
        for (int q = 0; q < 4; ++q) {
            bool valid = (cy[q] >= 0.f) && (cy[q] <= (float)(Himg-1)) &&
                         (cx[q] >= 0.f) && (cx[q] <= (float)(Wimg-1));
            int yi = min(max((int)cy[q], 0), Himg-1);
            int xi = min(max((int)cx[q], 0), Wimg-1);
            ci[q] = yi*Wimg + xi;
            cw[q] = valid ? cw[q] : 0.0f;
        }
        midx[p*KTAP + k] = make_int4(ci[0], ci[1], ci[2], ci[3]);
        mwgt[p*KTAP + k] = make_float4(cw[0], cw[1], cw[2], cw[3]);
    }
}

// ---------------- Kernel 2: pack w_dcn into f16 B-fragment layout ----------------
// NEW layout: Bpack[kc][nt][lane][e] -> each K-chunk is a contiguous 16KB block.
//   N = lane%16 ; K_in_chunk = (lane>=16 ? 16 : 0) + e
__global__ void k_packB(const float* __restrict__ w_dcn, _Float16* __restrict__ Bpack)
{
    int tid  = blockIdx.x * 256 + threadIdx.x;     // 72*16*32*16 = 589824
    int e    = tid & 15;
    int lane = (tid >> 4) & 31;
    int nt   = (tid >> 9) & 15;
    int kc   = tid >> 13;
    int tap   = kc >> 3;
    int cbase = (kc & 7) * 32;
    int kin   = ((lane >= 16) ? 16 : 0) + e;
    int c     = cbase + kin;
    int o     = nt*16 + (lane & 15);
    Bpack[tid] = (_Float16)w_dcn[((size_t)o * C + c) * 9 + tap];
}

// ---------------- Kernel 3: WMMA deformable-conv GEMM ----------------
// 8 waves/block, 1 wave = 16 pixels x 256 out-channels (all 16 n-tiles).
// B is shared by every wave: stream each 16KB K-chunk into double-buffered LDS
// with global_load_async_to_lds_b128 (ASYNCcnt), overlap with the bilinear
// im2col A-fragment build, and feed the 16 WMMAs from LDS.
__global__ void __launch_bounds__(256)
k_dcn_wmma(const float* __restrict__ x,
           const _Float16* __restrict__ Bpack,
           const int4*  __restrict__ midx,
           const float4* __restrict__ mwgt,
           float* __restrict__ gout)
{
    __shared__ __align__(16) _Float16 sB[2*CHUNKH];   // 2 x 16KB

    int tid  = threadIdx.x;
    int wave = blockIdx.x * 8 + (tid >> 5);  // 0 .. MT-1  (= m-tile)
    int lane = tid & 31;
    int mt   = wave;

    int pm  = lane & 15;                 // A-matrix M index
    int pix = mt*16 + pm;
    int n   = pix >> 12;
    const float* xn = x + (size_t)n * C * HW;
    int hiK = (lane >= 16) ? 16 : 0;     // A-layout lane-half K offset

    unsigned sbase = (unsigned)(uintptr_t)(&sB[0]);   // LDS byte offset

    // prologue: async-copy K-chunk 0 into buffer 0
    {
        const char* gp = (const char*)Bpack;
        #pragma unroll
        for (int l = 0; l < 4; ++l) {
            unsigned loff = (unsigned)(l*256 + tid) * 16u;
            unsigned la = sbase + loff;
            unsigned long long ga = (unsigned long long)(gp + loff);
            asm volatile("global_load_async_to_lds_b128 %0, %1, off"
                         :: "v"(la), "v"(ga) : "memory");
        }
    }
    asm volatile("s_wait_asynccnt 0x0" ::: "memory");
    __syncthreads();

    v8f acc[NTW];
    #pragma unroll
    for (int j = 0; j < NTW; ++j) acc[j] = (v8f){};

    for (int kc = 0; kc < NKC; ++kc) {
        int cur = kc & 1;

        // issue async copy of next K-chunk into the other buffer
        if (kc + 1 < NKC) {
            const char* gp = (const char*)(Bpack + (size_t)(kc + 1) * CHUNKH);
            unsigned lbase = sbase + (unsigned)((kc + 1) & 1) * (CHUNKH*2);
            #pragma unroll
            for (int l = 0; l < 4; ++l) {
                unsigned loff = (unsigned)(l*256 + tid) * 16u;
                unsigned la = lbase + loff;
                unsigned long long ga = (unsigned long long)(gp + loff);
                asm volatile("global_load_async_to_lds_b128 %0, %1, off"
                             :: "v"(la), "v"(ga) : "memory");
            }
        }

        int tap   = kc >> 3;
        int cbase = (kc & 7) * 32;
        int4   mi = midx[pix*KTAP + tap];
        float4 mw = mwgt[pix*KTAP + tap];

        // A fragment: K = 16*(lane/16) + e, matching Bpack (built once per kc)
        v16h afrag;
        #pragma unroll
        for (int e = 0; e < 16; ++e) {
            int v  = e >> 1;
            int h2 = e & 1;
            int kin = hiK + v*2 + h2;
            const float* pc = xn + (size_t)(cbase + kin) * HW;
            float val = mw.x * pc[mi.x] + mw.y * pc[mi.y]
                      + mw.z * pc[mi.z] + mw.w * pc[mi.w];
            afrag[e] = (_Float16)val;
        }

        // 16 B fragments from LDS, 16 WMMAs amortizing the single A build
        const _Float16* bbuf = sB + cur*CHUNKH;
        #pragma unroll
        for (int j = 0; j < NTW; ++j) {
            v16h bfrag = *(const v16h*)(bbuf + (j*32 + lane)*16);
            acc[j] = __builtin_amdgcn_wmma_f32_16x16x32_f16(
                         false, afrag, false, bfrag, (short)0, acc[j], false, false);
        }

        // next buffer must be fully resident block-wide before next iteration
        asm volatile("s_wait_asynccnt 0x0" ::: "memory");
        __syncthreads();
    }

    // D layout: VGPR r -> lanes0-15: M=r, lanes16-31: M=r+8 ; N = lane%16
    #pragma unroll
    for (int j = 0; j < NTW; ++j) {
        int o = j*16 + (lane & 15);
        #pragma unroll
        for (int r = 0; r < 8; ++r) {
            int m2   = r + ((lane >= 16) ? 8 : 0);
            int pix2 = mt*16 + m2;
            int n2   = pix2 >> 12;
            int hw2  = pix2 & 4095;
            gout[((size_t)(n2*OC + o))*HW + hw2] = acc[j][r];
        }
    }
}

// ---------------- Kernel 4: GroupNorm stats (1 block per (n,group)) ----------------
__global__ void k_gnstats(const float* __restrict__ gout, float* __restrict__ stats)
{
    __shared__ float s1[256], s2[256];
    int bid = blockIdx.x;              // n*32 + g
    int tid = threadIdx.x;
    const float* p = gout + (size_t)bid * (8*HW);   // group = 8 contiguous channels
    float sum = 0.f, sq = 0.f;
    for (int i = tid; i < 8*HW; i += 256) { float v = p[i]; sum += v; sq += v*v; }
    s1[tid] = sum; s2[tid] = sq;
    __syncthreads();
    for (int s = 128; s > 0; s >>= 1) {
        if (tid < s) { s1[tid] += s1[tid+s]; s2[tid] += s2[tid+s]; }
        __syncthreads();
    }
    if (tid == 0) {
        float mu  = s1[0] * (1.0f/(8*HW));
        float var = s2[0] * (1.0f/(8*HW)) - mu*mu;
        stats[2*bid]   = mu;
        stats[2*bid+1] = rsqrtf(var + EPSV);
    }
}

// ---------------- Kernel 5: normalize + ReLU ----------------
__global__ void k_gnapply(const float* __restrict__ gout,
                          const float* __restrict__ stats,
                          const float* __restrict__ gamma,
                          const float* __restrict__ beta,
                          float* __restrict__ out)
{
    int idx = blockIdx.x * 256 + threadIdx.x;       // NB*OC*HW = 8,388,608
    int n = idx >> 20;
    int c = (idx >> 12) & 255;
    int sb = 2*(n*32 + (c >> 3));
    float mu = stats[sb], rs = stats[sb+1];
    float y = (gout[idx] - mu) * rs * gamma[c] + beta[c];
    out[idx] = fmaxf(y, 0.0f);
}

extern "C" void kernel_launch(void* const* d_in, const int* in_sizes, int n_in,
                              void* d_out, int out_size, void* d_ws, size_t ws_size,
                              hipStream_t stream) {
    const float* x      = (const float*)d_in[0];
    const float* w_tm   = (const float*)d_in[1];
    const float* b_tm   = (const float*)d_in[2];
    const float* w_dcn  = (const float*)d_in[3];
    const float* gamma  = (const float*)d_in[4];
    const float* beta   = (const float*)d_in[5];
    float* out = (float*)d_out;

    char* ws = (char*)d_ws;
    size_t off = 0;
    _Float16* Bpack = (_Float16*)(ws + off); off += (size_t)NKC*NT*32*16*sizeof(_Float16); // 1.18 MB
    int4*    midx   = (int4*)   (ws + off); off += (size_t)MTOT*KTAP*sizeof(int4);         // 4.72 MB
    float4*  mwgt   = (float4*) (ws + off); off += (size_t)MTOT*KTAP*sizeof(float4);       // 4.72 MB
    float*   gout   = (float*)  (ws + off); off += (size_t)NB*OC*HW*sizeof(float);         // 33.55 MB
    float*   stats  = (float*)  (ws + off); off += (size_t)NB*32*2*sizeof(float);

    k_offsets <<<MTOT/256, 256, 0, stream>>>(x, w_tm, b_tm, midx, mwgt);
    k_packB   <<<(NKC*NT*32*16)/256, 256, 0, stream>>>(w_dcn, Bpack);
    k_dcn_wmma<<<MT/8, 256, 0, stream>>>(x, Bpack, midx, mwgt, gout);
    k_gnstats <<<NB*32, 256, 0, stream>>>(gout, stats);
    k_gnapply <<<(NB*OC*HW)/256, 256, 0, stream>>>(gout, stats, gamma, beta, out);
}
